// AttentionSubsample_16965120819385
// MI455X (gfx1250) — matrix operand order repaired and checked by reading, not compiled, not run on hardware
//
#include <hip/hip_runtime.h>
#include <hip/hip_bf16.h>

// ---------------- problem constants ----------------
#define BB    64
#define NN_   784      // keys per batch
#define CC    512
#define HH    16
#define KDIM  32
#define DV    64
#define NQ    196
#define NQP   208      // padded queries (13*16)
#define NKP   800      // padded keys   (25*32)
#define MKV   (BB*NN_)   // 50176
#define MQ    (BB*NQ)    // 12544
#define OKV   (HH*(KDIM+DV)) // 1536
#define OQ    (HH*KDIM)      // 512
#define OP    768
#define DH    (HH*DV)        // 1024
#define EPSBN 1e-5f
#define QKSCALE 0.17677669529663689f  // 1/sqrt(32)

typedef __bf16 bf16x16 __attribute__((ext_vector_type(16)));
typedef __bf16 bf16x8  __attribute__((ext_vector_type(8)));
typedef float  f32x8   __attribute__((ext_vector_type(8)));

// A-fragment (16x32 bf16, M x K): lane row = lane&15; half = lane>>4.
// half 0 -> K[0..15]; half 1 -> K[16..31]
static __device__ inline bf16x16 load_afrag(const __bf16* rowp, int half) {
  bf16x8 lo = *(const bf16x8*)(rowp + 16 * half);
  bf16x8 hi = *(const bf16x8*)(rowp + 16 * half + 8);
  bf16x16 r;
#pragma unroll
  for (int i = 0; i < 8; ++i) { r[i] = lo[i]; r[i + 8] = hi[i]; }
  return r;
}

static __device__ inline f32x8 wmma_bf16(bf16x16 a, bf16x16 b, f32x8 c) {
  return __builtin_amdgcn_wmma_f32_16x16x32_bf16(false, a, false, b, (short)0, c,
                                                 false, false);
}

// ---------------- utility kernels ----------------
__global__ void k_cvt_bf16(const float* __restrict__ in, __bf16* __restrict__ out,
                           size_t n) {
  size_t i = blockIdx.x * (size_t)blockDim.x + threadIdx.x;
  size_t st = (size_t)gridDim.x * blockDim.x;
  for (; i < n; i += st) out[i] = (__bf16)in[i];
}

__global__ void k_fill_bf16(__bf16* p, size_t n) {
  size_t i = blockIdx.x * (size_t)blockDim.x + threadIdx.x;
  size_t st = (size_t)gridDim.x * blockDim.x;
  for (; i < n; i += st) p[i] = (__bf16)0.0f;
}

__global__ void k_fill_f32(float* p, size_t n) {
  size_t i = blockIdx.x * (size_t)blockDim.x + threadIdx.x;
  size_t st = (size_t)gridDim.x * blockDim.x;
  for (; i < n; i += st) p[i] = 0.0f;
}

// xq = strided subsample of x, converted to bf16.  [MQ][CC]
__global__ void k_gather_xq(const float* __restrict__ x, __bf16* __restrict__ xq) {
  size_t n = (size_t)MQ * CC;
  size_t i = blockIdx.x * (size_t)blockDim.x + threadIdx.x;
  size_t st = (size_t)gridDim.x * blockDim.x;
  for (; i < n; i += st) {
    int c  = (int)(i % CC);
    int qi = (int)((i / CC) % NQ);
    int b  = (int)(i / ((size_t)CC * NQ));
    int r  = (qi / 14) * 2, cl = (qi % 14) * 2;
    int nidx = r * 28 + cl;
    xq[i] = (__bf16)x[((size_t)b * NN_ + nidx) * CC + c];
  }
}

// Bias[h][qp][kp]  (padded; -1e30 outside valid range)
__global__ void k_bias_expand(const float* __restrict__ ab,
                              const int* __restrict__ idxs, float* __restrict__ bias,
                              int n_off) {
  size_t n = (size_t)HH * NQP * NKP;
  size_t i = blockIdx.x * (size_t)blockDim.x + threadIdx.x;
  size_t st = (size_t)gridDim.x * blockDim.x;
  for (; i < n; i += st) {
    int kp = (int)(i % NKP);
    int qp = (int)((i / NKP) % NQP);
    int h  = (int)(i / ((size_t)NKP * NQP));
    float v = -1e30f;
    if (qp < NQ && kp < NN_) v = ab[h * n_off + idxs[qp * NN_ + kp]];
    bias[i] = v;
  }
}

// column sum / sumsq with fp32 atomics; grid (ceil(N/256), 64)
template <typename T>
__global__ void k_col_stats(const T* __restrict__ Y, int M, int N,
                            float* __restrict__ sum, float* __restrict__ sq) {
  int c = blockIdx.x * blockDim.x + threadIdx.x;
  if (c >= N) return;
  int chunk = (M + gridDim.y - 1) / gridDim.y;
  int r0 = blockIdx.y * chunk;
  int r1 = r0 + chunk; if (r1 > M) r1 = M;
  float s = 0.f, q = 0.f;
  for (int r = r0; r < r1; ++r) {
    float v = (float)Y[(size_t)r * N + c];
    s += v; q += v * v;
  }
  atomicAdd(&sum[c], s);
  atomicAdd(&sq[c], q);
}

static __device__ inline void bn_coef(const float* sum, const float* sq,
                                      const float* g, const float* bb, int o,
                                      float invM, float* sc, float* sh) {
  float mean = sum[o] * invM;
  float var  = sq[o] * invM - mean * mean;
  float s = g[o] * rsqrtf(var + EPSBN);
  *sc = s;
  *sh = bb[o] - mean * s;
}

// Ykv[MKV][OKV] -> K[b*H+h][key(800)][32], Vt[b*H+h][dim(64)][key(800)]
__global__ void k_bn_kv(const __bf16* __restrict__ Y, const float* __restrict__ sum,
                        const float* __restrict__ sq, const float* __restrict__ g,
                        const float* __restrict__ bb, __bf16* __restrict__ Kb,
                        __bf16* __restrict__ Vt) {
  size_t n = (size_t)MKV * OKV;
  size_t i = blockIdx.x * (size_t)blockDim.x + threadIdx.x;
  size_t st = (size_t)gridDim.x * blockDim.x;
  const float invM = 1.0f / MKV;
  for (; i < n; i += st) {
    int o = (int)(i % OKV);
    size_t row = i / OKV;
    int key = (int)(row % NN_);
    int b   = (int)(row / NN_);
    float sc, sh;
    bn_coef(sum, sq, g, bb, o, invM, &sc, &sh);
    float v = (float)Y[i] * sc + sh;
    int h = o / (KDIM + DV), j = o % (KDIM + DV);
    size_t bh = (size_t)b * HH + h;
    if (j < KDIM) Kb[(bh * NKP + key) * KDIM + j] = (__bf16)v;
    else          Vt[(bh * DV + (j - KDIM)) * NKP + key] = (__bf16)v;
  }
}

// Yq[MQ][OQ] -> Qb[b*H+h][q(208)][32]; folds 1/sqrt(KD) into the BN affine
__global__ void k_bn_q(const __bf16* __restrict__ Y, const float* __restrict__ sum,
                       const float* __restrict__ sq, const float* __restrict__ g,
                       const float* __restrict__ bb, __bf16* __restrict__ Qb) {
  size_t n = (size_t)MQ * OQ;
  size_t i = blockIdx.x * (size_t)blockDim.x + threadIdx.x;
  size_t st = (size_t)gridDim.x * blockDim.x;
  const float invM = 1.0f / MQ;
  for (; i < n; i += st) {
    int o = (int)(i % OQ);
    size_t row = i / OQ;
    int qi = (int)(row % NQ);
    int b  = (int)(row / NQ);
    float sc, sh;
    bn_coef(sum, sq, g, bb, o, invM, &sc, &sh);
    float v = (float)Y[i] * (sc * QKSCALE) + (sh * QKSCALE);
    int h = o / KDIM, d = o % KDIM;
    Qb[(((size_t)b * HH + h) * NQP + qi) * KDIM + d] = (__bf16)v;
  }
}

// Yp[MQ][OP] f32 -> out (BN applied)
__global__ void k_bn_proj(const float* __restrict__ Y, const float* __restrict__ sum,
                          const float* __restrict__ sq, const float* __restrict__ g,
                          const float* __restrict__ bb, float* __restrict__ out) {
  size_t n = (size_t)MQ * OP;
  size_t i = blockIdx.x * (size_t)blockDim.x + threadIdx.x;
  size_t st = (size_t)gridDim.x * blockDim.x;
  const float invM = 1.0f / MQ;
  for (; i < n; i += st) {
    int o = (int)(i % OP);
    float sc, sh;
    bn_coef(sum, sq, g, bb, o, invM, &sc, &sh);
    out[i] = Y[i] * sc + sh;
  }
}

// ---------------- WMMA GEMM:  C[M][N] = A[M][K] * B[N][K]^T ----------------
// 256 threads = 8 waves; wave tile 16x128 (8 WMMAs / A-fragment); block 128x128.
// M % 128 == 0, N % 128 == 0, K % 32 == 0 (true for all three uses).
template <bool OUT_BF16>
__global__ __launch_bounds__(256) void k_gemm(const __bf16* __restrict__ A,
                                              const __bf16* __restrict__ Bm,
                                              void* __restrict__ Cout, int M, int N,
                                              int K) {
  int wave = threadIdx.x >> 5, lane = threadIdx.x & 31;
  int nn = lane & 15, half = lane >> 4;
  int m0 = blockIdx.y * 128 + wave * 16;
  int n0 = blockIdx.x * 128;
  const __bf16* arow = A + (size_t)(m0 + nn) * K;
  const __bf16* brow = Bm + (size_t)(n0 + nn) * K + 16 * half;
  const size_t bstep = (size_t)16 * K;
  f32x8 acc[8];
#pragma unroll
  for (int j = 0; j < 8; ++j) acc[j] = (f32x8){};
  for (int kk = 0; kk < K; kk += 32) {
    __builtin_prefetch(arow + kk + 256, 0, 1);
    bf16x16 af = load_afrag(arow + kk, half);
#pragma unroll
    for (int j = 0; j < 8; ++j) {
      bf16x16 bf = *(const bf16x16*)(brow + (size_t)j * bstep + kk);
      acc[j] = wmma_bf16(af, bf, acc[j]);
    }
  }
#pragma unroll
  for (int r = 0; r < 8; ++r) {
    int m = m0 + r + 8 * half;
    size_t base = (size_t)m * N + n0 + nn;
    if (OUT_BF16) {
      __bf16* C = (__bf16*)Cout;
#pragma unroll
      for (int j = 0; j < 8; ++j) C[base + 16 * j] = (__bf16)acc[j][r];
    } else {
      float* C = (float*)Cout;
#pragma unroll
      for (int j = 0; j < 8; ++j) C[base + 16 * j] = acc[j][r];
    }
  }
}

// ---------------- fused flash attention ----------------
// grid = B*H; block = 13 waves (416 threads); wave w handles query tile w.
// Row sums computed with an extra WMMA against an all-ones B fragment.
__global__ __launch_bounds__(416) void k_attn(const __bf16* __restrict__ Qb,
                                              const __bf16* __restrict__ Kb,
                                              const __bf16* __restrict__ Vt,
                                              const float* __restrict__ Bias,
                                              __bf16* __restrict__ Hsw) {
  int bh = blockIdx.x;
  int b = bh >> 4, h = bh & 15;
  int wave = threadIdx.x >> 5, lane = threadIdx.x & 31;
  int nn = lane & 15, half = lane >> 4;
  int q0 = wave * 16;
  __shared__ __attribute__((aligned(16))) __bf16 pl[13 * 16 * 40];
  __bf16* myP = &pl[wave * 16 * 40];

  const __bf16* kBase = Kb + (size_t)bh * NKP * KDIM;
  const __bf16* vBase = Vt + (size_t)bh * DV * NKP;
  const float* biasBase = Bias + ((size_t)h * NQP + q0) * NKP;

  bf16x16 qf = load_afrag(Qb + ((size_t)bh * NQP + q0 + nn) * KDIM, half);
  bf16x16 ones;
#pragma unroll
  for (int i = 0; i < 16; ++i) ones[i] = (__bf16)1.0f;

  f32x8 o0 = {}, o1 = {}, o2 = {}, o3 = {};
  float mrow[8], lrow[8];
#pragma unroll
  for (int r = 0; r < 8; ++r) { mrow[r] = -1e30f; lrow[r] = 0.0f; }

  for (int t = 0; t < NKP / 32; ++t) {
    int key0 = t * 32;
    // QK^T : two 16-key B-fragments (contiguous 32B per lane); scale pre-folded in Q
    bf16x16 kf0 = *(const bf16x16*)(kBase + (size_t)(key0 + nn) * KDIM + 16 * half);
    bf16x16 kf1 = *(const bf16x16*)(kBase + (size_t)(key0 + 16 + nn) * KDIM + 16 * half);
    f32x8 z = {};
    f32x8 s0 = wmma_bf16(qf, kf0, z);
    f32x8 s1 = wmma_bf16(qf, kf1, z);

    float alpha[8];
#pragma unroll
    for (int r = 0; r < 8; ++r) {
      int qr = r + 8 * half;
      float v0 = s0[r] + biasBase[(size_t)qr * NKP + key0 + nn];
      float v1 = s1[r] + biasBase[(size_t)qr * NKP + key0 + 16 + nn];
      float mt = fmaxf(v0, v1);
      mt = fmaxf(mt, __shfl_xor(mt, 1, 32));
      mt = fmaxf(mt, __shfl_xor(mt, 2, 32));
      mt = fmaxf(mt, __shfl_xor(mt, 4, 32));
      mt = fmaxf(mt, __shfl_xor(mt, 8, 32));
      float mn = fmaxf(mrow[r], mt);
      alpha[r] = __expf(mrow[r] - mn);
      mrow[r] = mn;
      // write P tile (C-layout -> LDS) for transpose to A-layout
      myP[qr * 40 + nn]      = (__bf16)__expf(v0 - mn);
      myP[qr * 40 + 16 + nn] = (__bf16)__expf(v1 - mn);
    }
    bf16x16 pf = load_afrag(&myP[nn * 40], half);
    // row sums of (bf16-rounded) P via WMMA with ones
    f32x8 z2 = {};
    f32x8 srow = wmma_bf16(pf, ones, z2);
    // P * V : four 16-dim B-fragments from transposed V
    bf16x16 vf0 = *(const bf16x16*)(vBase + (size_t)(0 + nn) * NKP + key0 + 16 * half);
    bf16x16 vf1 = *(const bf16x16*)(vBase + (size_t)(16 + nn) * NKP + key0 + 16 * half);
    bf16x16 vf2 = *(const bf16x16*)(vBase + (size_t)(32 + nn) * NKP + key0 + 16 * half);
    bf16x16 vf3 = *(const bf16x16*)(vBase + (size_t)(48 + nn) * NKP + key0 + 16 * half);
#pragma unroll
    for (int r = 0; r < 8; ++r) {
      lrow[r] = lrow[r] * alpha[r] + srow[r];
      o0[r] *= alpha[r]; o1[r] *= alpha[r];
      o2[r] *= alpha[r]; o3[r] *= alpha[r];
    }
    o0 = wmma_bf16(pf, vf0, o0);
    o1 = wmma_bf16(pf, vf1, o1);
    o2 = wmma_bf16(pf, vf2, o2);
    o3 = wmma_bf16(pf, vf3, o3);
  }

#pragma unroll
  for (int r = 0; r < 8; ++r) {
    int qq = q0 + r + 8 * half;
    if (qq < NQ) {
      float rl = 1.0f / lrow[r];
      __bf16* orow = Hsw + ((size_t)b * NQ + qq) * DH + h * DV;
      float v;
      v = o0[r] * rl; orow[nn]      = (__bf16)(v * fminf(fmaxf(v + 3.f, 0.f), 6.f) * (1.f / 6.f));
      v = o1[r] * rl; orow[16 + nn] = (__bf16)(v * fminf(fmaxf(v + 3.f, 0.f), 6.f) * (1.f / 6.f));
      v = o2[r] * rl; orow[32 + nn] = (__bf16)(v * fminf(fmaxf(v + 3.f, 0.f), 6.f) * (1.f / 6.f));
      v = o3[r] * rl; orow[48 + nn] = (__bf16)(v * fminf(fmaxf(v + 3.f, 0.f), 6.f) * (1.f / 6.f));
    }
  }
}

// ---------------- host launcher ----------------
static inline int gsz(size_t n) {
  size_t b = (n + 255) / 256;
  return (int)(b < 4096 ? b : 4096);
}

extern "C" void kernel_launch(void* const* d_in, const int* in_sizes, int n_in,
                              void* d_out, int out_size, void* d_ws, size_t ws_size,
                              hipStream_t stream) {
  const float* x      = (const float*)d_in[0];
  const float* W_kv   = (const float*)d_in[1];
  const float* g_kv   = (const float*)d_in[2];
  const float* b_kv   = (const float*)d_in[3];
  const float* W_q    = (const float*)d_in[4];
  const float* g_q    = (const float*)d_in[5];
  const float* b_q    = (const float*)d_in[6];
  const float* W_proj = (const float*)d_in[7];
  const float* g_proj = (const float*)d_in[8];
  const float* b_proj = (const float*)d_in[9];
  const float* ab     = (const float*)d_in[10];
  const int*   idxs   = (const int*)d_in[11];
  int n_off = in_sizes[10] / HH;

  // workspace carve
  char* w = (char*)d_ws;
  size_t off = 0;
  auto carve = [&](size_t bytes) {
    void* p = w + off;
    off = (off + bytes + 255) & ~(size_t)255;
    return p;
  };
  __bf16* xb   = (__bf16*)carve((size_t)MKV * CC * 2);
  __bf16* xqb  = (__bf16*)carve((size_t)MQ * CC * 2);
  __bf16* Wkvb = (__bf16*)carve((size_t)OKV * CC * 2);
  __bf16* Wqb  = (__bf16*)carve((size_t)OQ * CC * 2);
  __bf16* Wpb  = (__bf16*)carve((size_t)OP * DH * 2);
  __bf16* Ykv  = (__bf16*)carve((size_t)MKV * OKV * 2);
  __bf16* Yq   = (__bf16*)carve((size_t)MQ * OQ * 2);
  __bf16* Kb   = (__bf16*)carve((size_t)BB * HH * NKP * KDIM * 2);
  __bf16* Vt   = (__bf16*)carve((size_t)BB * HH * DV * NKP * 2);
  __bf16* Qb   = (__bf16*)carve((size_t)BB * HH * NQP * KDIM * 2);
  float*  Bias = (float*)carve((size_t)HH * NQP * NKP * 4);
  __bf16* Hsw  = (__bf16*)carve((size_t)MQ * DH * 2);
  float*  Yp   = (float*)carve((size_t)MQ * OP * 4);
  float*  stats = (float*)carve((size_t)(2 * OKV + 2 * OQ + 2 * OP) * 4);
  float* sum_kv = stats,            * sq_kv = stats + OKV;
  float* sum_q  = stats + 2 * OKV,  * sq_q  = stats + 2 * OKV + OQ;
  float* sum_p  = stats + 2 * OKV + 2 * OQ, * sq_p = sum_p + OP;

  // 1) converts / gathers / fills
  k_cvt_bf16<<<gsz((size_t)MKV * CC), 256, 0, stream>>>(x, xb, (size_t)MKV * CC);
  k_cvt_bf16<<<gsz((size_t)OKV * CC), 256, 0, stream>>>(W_kv, Wkvb, (size_t)OKV * CC);
  k_cvt_bf16<<<gsz((size_t)OQ * CC), 256, 0, stream>>>(W_q, Wqb, (size_t)OQ * CC);
  k_cvt_bf16<<<gsz((size_t)OP * DH), 256, 0, stream>>>(W_proj, Wpb, (size_t)OP * DH);
  k_gather_xq<<<gsz((size_t)MQ * CC), 256, 0, stream>>>(x, xqb);
  k_fill_bf16<<<gsz((size_t)BB * HH * NKP * KDIM), 256, 0, stream>>>(
      Kb, (size_t)BB * HH * NKP * KDIM);
  k_fill_bf16<<<gsz((size_t)BB * HH * DV * NKP), 256, 0, stream>>>(
      Vt, (size_t)BB * HH * DV * NKP);
  k_fill_bf16<<<gsz((size_t)BB * HH * NQP * KDIM), 256, 0, stream>>>(
      Qb, (size_t)BB * HH * NQP * KDIM);
  k_fill_f32<<<1, 256, 0, stream>>>(stats, (size_t)(2 * OKV + 2 * OQ + 2 * OP));
  k_bias_expand<<<gsz((size_t)HH * NQP * NKP), 256, 0, stream>>>(ab, idxs, Bias, n_off);

  // 2) kv and q GEMMs (bf16 out) + stats + BN scatter
  {
    dim3 g(OKV / 128, MKV / 128);
    k_gemm<true><<<g, 256, 0, stream>>>(xb, Wkvb, Ykv, MKV, OKV, CC);
  }
  {
    dim3 g(OQ / 128, MQ / 128);
    k_gemm<true><<<g, 256, 0, stream>>>(xqb, Wqb, Yq, MQ, OQ, CC);
  }
  {
    dim3 g((OKV + 255) / 256, 64);
    k_col_stats<__bf16><<<g, 256, 0, stream>>>(Ykv, MKV, OKV, sum_kv, sq_kv);
  }
  {
    dim3 g((OQ + 255) / 256, 64);
    k_col_stats<__bf16><<<g, 256, 0, stream>>>(Yq, MQ, OQ, sum_q, sq_q);
  }
  k_bn_kv<<<gsz((size_t)MKV * OKV), 256, 0, stream>>>(Ykv, sum_kv, sq_kv, g_kv, b_kv,
                                                      Kb, Vt);
  k_bn_q<<<gsz((size_t)MQ * OQ), 256, 0, stream>>>(Yq, sum_q, sq_q, g_q, b_q, Qb);

  // 3) fused attention + hard-swish
  k_attn<<<BB * HH, 416, 0, stream>>>(Qb, Kb, Vt, Bias, Hsw);

  // 4) proj GEMM (f32 out) + stats + BN -> d_out
  {
    dim3 g(OP / 128, MQ / 128);
    k_gemm<false><<<g, 256, 0, stream>>>(Hsw, Wpb, Yp, MQ, OP, DH);
  }
  {
    dim3 g((OP + 255) / 256, 64);
    k_col_stats<float><<<g, 256, 0, stream>>>(Yp, MQ, OP, sum_p, sq_p);
  }
  k_bn_proj<<<gsz((size_t)MQ * OP), 256, 0, stream>>>(Yp, sum_p, sq_p, g_proj, b_proj,
                                                      (float*)d_out);
}
